// Model_34110630264968
// MI455X (gfx1250) — matrix operand order repaired and checked by reading, not compiled, not run on hardware
//
#include <hip/hip_runtime.h>
#include <hip/hip_bf16.h>
#include <math.h>

// ---------------------------------------------------------------------------
// Model constants (from reference)
// ---------------------------------------------------------------------------
#define BSZ   32
#define SEQL  512
#define PLEN  32          // P
#define NPAT  16          // SEQ / P
#define MPAT  4           // M (AR patches)
#define DMODEL 1024       // D
#define DFF3   3072
#define DFFN   4096
#define NH    16          // H
#define HDIM  64          // HD
#define CMAX  20          // max KV length (16 prefill + 4 AR)
#define QDIM  10          // Q
#define ODIM  128         // O
#define OSDIM 256         // OS
#define HOR   256
#define EPSV  1e-6f

// ---------------------------------------------------------------------------
// WMMA types / helpers
// ---------------------------------------------------------------------------
typedef __attribute__((ext_vector_type(16))) __bf16 v16bf;
typedef __attribute__((ext_vector_type(8)))  float  v8f;

union Frag { uint4 u[2]; v16bf v; };

__device__ __forceinline__ __bf16 f2bf(float f) {
  unsigned u = __float_as_uint(f);
  unsigned r = (u + 0x7FFFu + ((u >> 16) & 1u)) >> 16;   // round-to-nearest-even
  union { unsigned short us; __bf16 h; } cv;
  cv.us = (unsigned short)r;
  return cv.h;
}

__device__ __forceinline__ float waveReduce(float v) {
  #pragma unroll
  for (int off = 16; off > 0; off >>= 1) v += __shfl_xor(v, off, 32);
  return v;
}

__device__ __forceinline__ float gelu_f(float x) {
  float x3 = x * x * x;
  return 0.5f * x * (1.f + tanhf(0.7978845608028654f * (x + 0.044715f * x3)));
}

// --- async global -> LDS 16B copy (CDNA5 GLOBAL_LOAD_ASYNC_TO_LDS_B128) -----
// VDST VGPR = 32-bit wave-relative LDS byte address; generic __shared__
// pointers carry that address in their low 32 bits (ISA 10.2 aperture rules).
__device__ __forceinline__ void copy16_g2lds(const __bf16* g, __bf16* l) {
  unsigned lds_addr = (unsigned)(unsigned long long)l;
  asm volatile("global_load_async_to_lds_b128 %0, %1, off"
               :
               : "v"(lds_addr), "v"((unsigned long long)g)
               : "memory");
}

__device__ __forceinline__ void async_wait0() {
#if __has_builtin(__builtin_amdgcn_s_wait_asynccnt)
  __builtin_amdgcn_s_wait_asynccnt(0);
#else
  asm volatile("s_wait_asynccnt 0" ::: "memory");
#endif
}

// ---------------------------------------------------------------------------
// GEMM v2 (cached path): A[M,K] f32, Bt[N,K] bf16 (pre-transposed weights).
// Block tile 128x128, 8 waves; each wave 32x64 = 2x4 v_wmma_f32_16x16x32_bf16.
// K % 32 == 0, N % 128 == 0.  flags: 1=+bias, 2=+resid, 4=gelu
// ---------------------------------------------------------------------------
#define TK 32
#define AST (TK + 8)   // LDS row stride in bf16 elems (80B, 16B-aligned)
#define BST (TK + 8)

__global__ __launch_bounds__(256)
void gemm_bf16T(const float* __restrict__ A, const __bf16* __restrict__ Bt,
                const float* __restrict__ bias, const float* resid,
                float* __restrict__ Cout, int Mrows, int N, int K, int flags)
{
  __shared__ __align__(16) __bf16 As[128 * AST];   // A tile row-major [m][k]
  __shared__ __align__(16) __bf16 Bs[128 * BST];   // B tile [n][k]

  const int tid  = threadIdx.x;
  const int lane = tid & 31;
  const int wave = tid >> 5;
  const int wm   = wave >> 1;      // 0..3 : M sub-block of 32
  const int wn   = wave & 1;       // 0..1 : N sub-block of 64
  const int m0   = blockIdx.y * 128;
  const int n0   = blockIdx.x * 128;
  const int mlo  = lane & 15;
  const int hi   = lane >> 4;

  v8f acc[2][4];
  #pragma unroll
  for (int i = 0; i < 2; ++i)
    #pragma unroll
    for (int j = 0; j < 4; ++j)
      #pragma unroll
      for (int r = 0; r < 8; ++r) acc[i][j][r] = 0.f;

  const int ksteps = K / TK;
  for (int kt = 0; kt < ksteps; ++kt) {
    const int k0 = kt * TK;

    // --- stage B tile: 128 rows x 64B of bf16, async 16B copies -----------
    #pragma unroll
    for (int i = 0; i < 2; ++i) {
      int idx = tid + i * 256;          // 512 chunks of 16B
      int n   = idx >> 2;
      int kc  = idx & 3;
      copy16_g2lds(Bt + (size_t)(n0 + n) * K + k0 + kc * 8,
                   &Bs[n * BST + kc * 8]);
    }
    // --- stage A tile: 128x32 f32 -> bf16 ---------------------------------
    #pragma unroll
    for (int i = 0; i < 4; ++i) {
      int idx = tid + i * 256;          // float4 index, 8 per row
      int r   = idx >> 3;
      int c4  = idx & 7;
      float4 v = make_float4(0.f, 0.f, 0.f, 0.f);
      int gr = m0 + r;
      if (gr < Mrows) v = *(const float4*)(A + (size_t)gr * K + k0 + c4 * 4);
      __bf16* dst = &As[r * AST + c4 * 4];
      dst[0] = f2bf(v.x); dst[1] = f2bf(v.y); dst[2] = f2bf(v.z); dst[3] = f2bf(v.w);
    }
    async_wait0();
    __syncthreads();

    // --- prefetch next K tile --------------------------------------------
    if (kt + 1 < ksteps) {
      int nk = k0 + TK;
      int ar = m0 + (tid >> 1);
      if (ar < Mrows)
        __builtin_prefetch(A + (size_t)ar * K + nk + (tid & 1) * 16, 0, 1);
      __builtin_prefetch(Bt + (size_t)(n0 + ((tid >> 1) & 127)) * K + nk, 0, 1);
    }

    // --- 2x4 WMMA per wave ------------------------------------------------
    #pragma unroll
    for (int i = 0; i < 2; ++i) {
      Frag fa;
      const __bf16* arow = &As[(wm * 32 + i * 16 + mlo) * AST];
      fa.u[0] = *(const uint4*)(arow + hi * 8);        // K {0..7}/{8..15}
      fa.u[1] = *(const uint4*)(arow + 16 + hi * 8);   // K {16..23}/{24..31}
      #pragma unroll
      for (int j = 0; j < 4; ++j) {
        Frag fb;
        const __bf16* brow = &Bs[(wn * 64 + j * 16 + mlo) * BST];
        fb.u[0] = *(const uint4*)(brow + hi * 8);
        fb.u[1] = *(const uint4*)(brow + 16 + hi * 8);
        acc[i][j] = __builtin_amdgcn_wmma_f32_16x16x32_bf16(
            false, fa.v, false, fb.v, (short)0, acc[i][j], false, false);
      }
    }
    __syncthreads();
  }

  const int nn = lane & 15;
  #pragma unroll
  for (int i = 0; i < 2; ++i) {
    #pragma unroll
    for (int j = 0; j < 4; ++j) {
      int gcol = n0 + wn * 64 + j * 16 + nn;
      float bv = (flags & 1) ? bias[gcol] : 0.f;
      #pragma unroll
      for (int r = 0; r < 8; ++r) {
        int grow = m0 + wm * 32 + i * 16 + hi * 8 + r;
        if (grow < Mrows) {
          float v = acc[i][j][r] + bv;
          if (flags & 4) v = gelu_f(v);
          if (flags & 2) v += resid[(size_t)grow * N + gcol];
          Cout[(size_t)grow * N + gcol] = v;
        }
      }
    }
  }
}

// ---------------------------------------------------------------------------
// GEMM fallback (small-ws path): B is f32 [K,N].  Block tile 128x64,
// 8 waves, each wave 2x2 WMMA.  N % 64 == 0, K % 32 == 0.
// ---------------------------------------------------------------------------
__global__ __launch_bounds__(256)
void gemm_f32B(const float* __restrict__ A, const float* __restrict__ Bw,
               const float* __restrict__ bias, const float* resid,
               float* __restrict__ Cout, int Mrows, int N, int K, int flags)
{
  __shared__ __align__(16) __bf16 As[128 * AST];
  __shared__ __align__(16) __bf16 Bs[64 * BST];    // [n][k]

  const int tid  = threadIdx.x;
  const int lane = tid & 31;
  const int wave = tid >> 5;
  const int wm   = wave >> 1;
  const int wn   = wave & 1;
  const int m0   = blockIdx.y * 128;
  const int n0   = blockIdx.x * 64;
  const int mlo  = lane & 15;
  const int hi   = lane >> 4;

  v8f acc[2][2];
  #pragma unroll
  for (int i = 0; i < 2; ++i)
    #pragma unroll
    for (int j = 0; j < 2; ++j)
      #pragma unroll
      for (int r = 0; r < 8; ++r) acc[i][j][r] = 0.f;

  const int ksteps = K / TK;
  for (int kt = 0; kt < ksteps; ++kt) {
    const int k0 = kt * TK;
    #pragma unroll
    for (int i = 0; i < 4; ++i) {
      int idx = tid + i * 256;
      int r   = idx >> 3;
      int c4  = idx & 7;
      float4 v = make_float4(0.f, 0.f, 0.f, 0.f);
      int gr = m0 + r;
      if (gr < Mrows) v = *(const float4*)(A + (size_t)gr * K + k0 + c4 * 4);
      __bf16* dst = &As[r * AST + c4 * 4];
      dst[0] = f2bf(v.x); dst[1] = f2bf(v.y); dst[2] = f2bf(v.z); dst[3] = f2bf(v.w);
    }
    #pragma unroll
    for (int i = 0; i < 2; ++i) {
      int idx = tid + i * 256;          // 512 float4, 16 per row of 64
      int r   = idx >> 4;
      int c4  = idx & 15;
      float4 v = *(const float4*)(Bw + (size_t)(k0 + r) * N + n0 + c4 * 4);
      Bs[(c4 * 4 + 0) * BST + r] = f2bf(v.x);
      Bs[(c4 * 4 + 1) * BST + r] = f2bf(v.y);
      Bs[(c4 * 4 + 2) * BST + r] = f2bf(v.z);
      Bs[(c4 * 4 + 3) * BST + r] = f2bf(v.w);
    }
    __syncthreads();
    if (kt + 1 < ksteps) {
      int nk = k0 + TK;
      int ar = m0 + (tid >> 1);
      if (ar < Mrows)
        __builtin_prefetch(A + (size_t)ar * K + nk + (tid & 1) * 16, 0, 1);
      __builtin_prefetch(Bw + (size_t)(nk + (tid & 31)) * N + n0 + ((tid >> 5) & 1) * 32, 0, 1);
    }
    #pragma unroll
    for (int i = 0; i < 2; ++i) {
      Frag fa;
      const __bf16* arow = &As[(wm * 32 + i * 16 + mlo) * AST];
      fa.u[0] = *(const uint4*)(arow + hi * 8);
      fa.u[1] = *(const uint4*)(arow + 16 + hi * 8);
      #pragma unroll
      for (int j = 0; j < 2; ++j) {
        Frag fb;
        const __bf16* brow = &Bs[(wn * 32 + j * 16 + mlo) * BST];
        fb.u[0] = *(const uint4*)(brow + hi * 8);
        fb.u[1] = *(const uint4*)(brow + 16 + hi * 8);
        acc[i][j] = __builtin_amdgcn_wmma_f32_16x16x32_bf16(
            false, fa.v, false, fb.v, (short)0, acc[i][j], false, false);
      }
    }
    __syncthreads();
  }
  const int nn = lane & 15;
  #pragma unroll
  for (int i = 0; i < 2; ++i) {
    #pragma unroll
    for (int j = 0; j < 2; ++j) {
      int gcol = n0 + wn * 32 + j * 16 + nn;
      float bv = (flags & 1) ? bias[gcol] : 0.f;
      #pragma unroll
      for (int r = 0; r < 8; ++r) {
        int grow = m0 + wm * 32 + i * 16 + hi * 8 + r;
        if (grow < Mrows) {
          float v = acc[i][j][r] + bv;
          if (flags & 4) v = gelu_f(v);
          if (flags & 2) v += resid[(size_t)grow * N + gcol];
          Cout[(size_t)grow * N + gcol] = v;
        }
      }
    }
  }
}

// ---------------------------------------------------------------------------
// Weight convert+transpose: W[K,N] f32 -> Wt[N,K] bf16.  32x32 LDS tile.
// ---------------------------------------------------------------------------
__global__ __launch_bounds__(256)
void convert_wT(const float* __restrict__ W, __bf16* __restrict__ Wt, int K, int N)
{
  __shared__ float tile[32][33];
  int kb = blockIdx.x * 32, nb = blockIdx.y * 32;
  int tx = threadIdx.x & 31, ty = threadIdx.x >> 5;   // ty 0..7
  #pragma unroll
  for (int i = ty; i < 32; i += 8)
    tile[i][tx] = W[(size_t)(kb + i) * N + nb + tx];
  __syncthreads();
  #pragma unroll
  for (int i = ty; i < 32; i += 8)
    Wt[(size_t)(nb + i) * K + kb + tx] = f2bf(tile[tx][i]);
}

// ---------------------------------------------------------------------------
// LayerNorm over D=1024, one block (256 thr) per row
// ---------------------------------------------------------------------------
__global__ __launch_bounds__(256)
void layernorm_k(const float* __restrict__ x, const float* __restrict__ s,
                 float* __restrict__ y)
{
  __shared__ float red[8];
  const int row = blockIdx.x;
  const float* xr = x + (size_t)row * DMODEL;
  float v[4];
  #pragma unroll
  for (int i = 0; i < 4; ++i) v[i] = xr[threadIdx.x + i * 256];
  float sum = v[0] + v[1] + v[2] + v[3];
  sum = waveReduce(sum);
  int wv = threadIdx.x >> 5, ln = threadIdx.x & 31;
  if (ln == 0) red[wv] = sum;
  __syncthreads();
  float tot = 0.f;
  #pragma unroll
  for (int i = 0; i < 8; ++i) tot += red[i];
  float mean = tot * (1.f / DMODEL);
  float var = 0.f;
  #pragma unroll
  for (int i = 0; i < 4; ++i) { float d = v[i] - mean; var += d * d; }
  var = waveReduce(var);
  __syncthreads();
  if (ln == 0) red[wv] = var;
  __syncthreads();
  float vt = 0.f;
  #pragma unroll
  for (int i = 0; i < 8; ++i) vt += red[i];
  float inv = rsqrtf(vt * (1.f / DMODEL) + EPSV);
  #pragma unroll
  for (int i = 0; i < 4; ++i) {
    int c = threadIdx.x + i * 256;
    y[(size_t)row * DMODEL + c] = (v[i] - mean) * inv * s[c];
  }
}

// ---------------------------------------------------------------------------
__global__ void append_kv(const float* __restrict__ qkv, float* __restrict__ Kc,
                          float* __restrict__ Vc, int Nn, int off)
{
  int t = blockIdx.x * 256 + threadIdx.x;
  int total = BSZ * Nn * DMODEL;
  if (t >= total) return;
  int d = t % DMODEL;
  int n = (t / DMODEL) % Nn;
  int b = t / (DMODEL * Nn);
  size_t src = (size_t)(b * Nn + n) * (3 * DMODEL);
  size_t dst = ((size_t)b * CMAX + off + n) * DMODEL + d;
  Kc[dst] = qkv[src + DMODEL + d];
  Vc[dst] = qkv[src + 2 * DMODEL + d];
}

// Causal attention, one wave per (b,h,n); lane covers d and d+32 of HD=64.
__global__ void attn_kernel(const float* __restrict__ qkv,
                            const float* __restrict__ Kc,
                            const float* __restrict__ Vc,
                            float* __restrict__ outp, int Nn, int off)
{
  int idx = blockIdx.x;
  int n = idx % Nn;
  int h = (idx / Nn) % NH;
  int b = idx / (Nn * NH);
  int lane = threadIdx.x;

  const float* q = qkv + (size_t)(b * Nn + n) * (3 * DMODEL) + h * HDIM;
  float q0 = q[lane], q1 = q[lane + 32];

  float sc[CMAX];
  int lim = n + off;
  float mx = -1e30f;
  for (int m = 0; m <= lim; ++m) {
    const float* kr = Kc + ((size_t)b * CMAX + m) * DMODEL + h * HDIM;
    float s = q0 * kr[lane] + q1 * kr[lane + 32];
    s = waveReduce(s) * 0.125f;
    sc[m] = s;
    mx = fmaxf(mx, s);
  }
  float den = 0.f;
  for (int m = 0; m <= lim; ++m) { sc[m] = __expf(sc[m] - mx); den += sc[m]; }
  float inv = 1.f / den;
  float o0 = 0.f, o1 = 0.f;
  for (int m = 0; m <= lim; ++m) {
    const float* vr = Vc + ((size_t)b * CMAX + m) * DMODEL + h * HDIM;
    o0 += sc[m] * vr[lane];
    o1 += sc[m] * vr[lane + 32];
  }
  float* orow = outp + (size_t)(b * Nn + n) * DMODEL + h * HDIM;
  orow[lane]      = o0 * inv;
  orow[lane + 32] = o1 * inv;
}

// ---------------------------------------------------------------------------
// Running stats (Welford per reference) + RevIN, prefill; 1 thread / series
// ---------------------------------------------------------------------------
__global__ void stats_prefill(const float* __restrict__ x, float sign,
                              float* __restrict__ normed, float* __restrict__ mus,
                              float* __restrict__ sigs, float* __restrict__ carry)
{
  int b = threadIdx.x;
  if (b >= BSZ) return;
  float n = 0.f, mu = 0.f, sg = 0.f;
  for (int p = 0; p < NPAT; ++p) {
    const float* xp = x + b * SEQL + p * PLEN;
    float sum = 0.f;
    for (int i = 0; i < PLEN; ++i) sum += sign * xp[i];
    float newn = n + (float)PLEN;
    float nmu = (n * mu + sum) / newn;
    float M2 = n * sg * sg;
    for (int i = 0; i < PLEN; ++i) { float v = sign * xp[i]; M2 += (v - mu) * (v - nmu); }
    float nsg = sqrtf(fmaxf(M2 / newn, 1e-12f));
    n = newn; mu = nmu; sg = nsg;
    mus[b * NPAT + p] = mu;
    sigs[b * NPAT + p] = sg;
    float inv = 1.f / (sg + EPSV);
    for (int i = 0; i < PLEN; ++i)
      normed[(b * NPAT + p) * PLEN + i] = (sign * xp[i] - mu) * inv;
  }
  carry[b * 3 + 0] = n;
  carry[b * 3 + 1] = mu;
  carry[b * 3 + 2] = sg;
}

__global__ void stats_ar(const float* __restrict__ arx, const float* __restrict__ carry,
                         float* __restrict__ nnorm, float* __restrict__ nmus,
                         float* __restrict__ nsigs)
{
  int b = threadIdx.x;
  if (b >= BSZ) return;
  float n = carry[b * 3], mu = carry[b * 3 + 1], sg = carry[b * 3 + 2];
  for (int p = 0; p < MPAT; ++p) {
    const float* xp = arx + b * ODIM + p * PLEN;
    float sum = 0.f;
    for (int i = 0; i < PLEN; ++i) sum += xp[i];
    float newn = n + (float)PLEN;
    float nmu = (n * mu + sum) / newn;
    float M2 = n * sg * sg;
    for (int i = 0; i < PLEN; ++i) M2 += (xp[i] - mu) * (xp[i] - nmu);
    float nsg = sqrtf(fmaxf(M2 / newn, 1e-12f));
    n = newn; mu = nmu; sg = nsg;
    nmus[b * MPAT + p] = mu;
    nsigs[b * MPAT + p] = sg;
    float inv = 1.f / (sg + EPSV);
    for (int i = 0; i < PLEN; ++i)
      nnorm[(b * MPAT + p) * PLEN + i] = (xp[i] - mu) * inv;
  }
}

__global__ void build_feat(const float* __restrict__ normed, float* __restrict__ feat,
                           int rows)
{
  int t = blockIdx.x * 256 + threadIdx.x;
  if (t >= rows * 64) return;
  int c = t & 63, r = t >> 6;
  feat[t] = (c < PLEN) ? normed[r * PLEN + c] : 0.f;
}

__global__ void revin_rows(const float* __restrict__ src, const float* __restrict__ mus,
                           const float* __restrict__ sigs, int patch, int npatch,
                           int rowlen, float* __restrict__ dst)
{
  int t = blockIdx.x * 256 + threadIdx.x;
  if (t >= BSZ * rowlen) return;
  int b = t / rowlen, c = t % rowlen;
  float mu = mus[b * npatch + patch];
  float sg = sigs[b * npatch + patch] + EPSV;
  dst[t] = src[(size_t)(b * npatch + patch) * rowlen + c] * sg + mu;
}

__global__ void extract_ar(const float* __restrict__ pf, float* __restrict__ arx)
{
  int t = blockIdx.x * 256 + threadIdx.x;
  if (t >= BSZ * ODIM) return;
  int b = t >> 7, o = t & 127;
  arx[t] = pf[b * (ODIM * QDIM) + o * QDIM + 5];
}

__global__ void ispos_k(const float* __restrict__ x, unsigned char* __restrict__ isp)
{
  int b = threadIdx.x;
  if (b >= BSZ) return;
  int ok = 1;
  for (int t = 0; t < SEQL; ++t) ok &= (x[b * SEQL + t] >= 0.f);
  isp[b] = (unsigned char)ok;
}

// ---------------------------------------------------------------------------
__device__ __forceinline__ int flipq(int q) { return q == 0 ? 0 : QDIM - q; }

__global__ void combine_k(const float* __restrict__ pfA, const float* __restrict__ arA,
                          const float* __restrict__ qsA, const float* __restrict__ pfB,
                          const float* __restrict__ arB, const float* __restrict__ qsB,
                          const unsigned char* __restrict__ isp, float* __restrict__ med)
{
  int t = blockIdx.x * 256 + threadIdx.x;
  if (t >= BSZ * HOR) return;
  int b = t >> 8, s = t & 255;

  float ff[QDIM], fff[QDIM], qc[QDIM];
  if (s < ODIM) {
    #pragma unroll
    for (int q = 0; q < QDIM; ++q) {
      ff[q]  = pfA[b * (ODIM * QDIM) + s * QDIM + q];
      fff[q] = pfB[b * (ODIM * QDIM) + s * QDIM + flipq(q)];  // flip_q(fpf)
    }
  } else {
    int o = s - ODIM;
    #pragma unroll
    for (int q = 0; q < QDIM; ++q) {
      ff[q]  = arA[b * (ODIM * QDIM) + o * QDIM + q];
      fff[q] = arB[b * (ODIM * QDIM) + o * QDIM + q];         // far NOT flipped
    }
  }
  #pragma unroll
  for (int q = 0; q < QDIM; ++q) {
    qc[q] = 0.5f * (qsA[b * (OSDIM * QDIM) + s * QDIM + q] -
                    qsB[b * (OSDIM * QDIM) + s * QDIM + flipq(q)]);
    ff[q] = 0.5f * (ff[q] - fff[q]);
  }
  float base = ff[5], q5 = qc[5];
  #pragma unroll
  for (int q = 1; q < QDIM; ++q)
    if (q != 5) ff[q] = qc[q] - q5 + base;
  for (int i = 4; i >= 1; --i) ff[i] = fminf(ff[i], ff[i + 1]);
  for (int i = 6; i <= 9; ++i) ff[i] = fmaxf(ff[i], ff[i - 1]);
  float m = ff[5];
  if (isp[b]) m = fmaxf(m, 0.f);
  med[t] = m;
}

// ---------------------------------------------------------------------------
// Host-side orchestration
// ---------------------------------------------------------------------------
struct WTb { const __bf16 *Win, *Wqkv, *Wo, *W1, *W2, *Wout, *Wqs; };

static inline void gemm(hipStream_t st, bool cached, const float* A,
                        const float* Bf, const __bf16* Bt,
                        const float* bias, const float* resid, float* C,
                        int M, int N, int K, int flags)
{
  if (cached) {
    dim3 g(N / 128, (M + 127) / 128), b(256);
    gemm_bf16T<<<g, b, 0, st>>>(A, Bt, bias, resid, C, M, N, K, flags);
  } else {
    dim3 g(N / 64, (M + 127) / 128), b(256);
    gemm_f32B<<<g, b, 0, st>>>(A, Bf, bias, resid, C, M, N, K, flags);
  }
}

static void run_model(hipStream_t st, bool cached, int Nn, int off,
                      const float* W_in, const float* b_in, const float* ln1,
                      const float* Wqkv, const float* Wo, const float* ln2,
                      const float* W1, const float* W2, const float* lnf,
                      const WTb& wt,
                      const float* feat, float* hdd, float* xn, float* qkvb,
                      float* attnb, float* ffnb, float* cacheK, float* cacheV,
                      float* hf)
{
  const int Mr = BSZ * Nn;
  gemm(st, cached, feat, W_in, wt.Win, b_in, nullptr, hdd, Mr, DMODEL, 64, 1);
  for (int l = 0; l < 8; ++l) {
    layernorm_k<<<Mr, 256, 0, st>>>(hdd, ln1 + l * DMODEL, xn);
    gemm(st, cached, xn, Wqkv + (size_t)l * DMODEL * DFF3,
         wt.Wqkv + (size_t)l * DMODEL * DFF3, nullptr, nullptr, qkvb,
         Mr, DFF3, DMODEL, 0);
    float* Kl = cacheK + (size_t)l * BSZ * CMAX * DMODEL;
    float* Vl = cacheV + (size_t)l * BSZ * CMAX * DMODEL;
    int tot = BSZ * Nn * DMODEL;
    append_kv<<<(tot + 255) / 256, 256, 0, st>>>(qkvb, Kl, Vl, Nn, off);
    attn_kernel<<<BSZ * NH * Nn, 32, 0, st>>>(qkvb, Kl, Vl, attnb, Nn, off);
    gemm(st, cached, attnb, Wo + (size_t)l * DMODEL * DMODEL,
         wt.Wo + (size_t)l * DMODEL * DMODEL, nullptr, hdd, hdd,
         Mr, DMODEL, DMODEL, 2);
    layernorm_k<<<Mr, 256, 0, st>>>(hdd, ln2 + l * DMODEL, xn);
    gemm(st, cached, xn, W1 + (size_t)l * DMODEL * DFFN,
         wt.W1 + (size_t)l * DMODEL * DFFN, nullptr, nullptr, ffnb,
         Mr, DFFN, DMODEL, 4);
    gemm(st, cached, ffnb, W2 + (size_t)l * DFFN * DMODEL,
         wt.W2 + (size_t)l * DFFN * DMODEL, nullptr, hdd, hdd,
         Mr, DMODEL, DFFN, 2);
  }
  layernorm_k<<<Mr, 256, 0, st>>>(hdd, lnf, hf);
}

extern "C" void kernel_launch(void* const* d_in, const int* in_sizes, int n_in,
                              void* d_out, int out_size, void* d_ws, size_t ws_size,
                              hipStream_t stream)
{
  (void)in_sizes; (void)n_in; (void)out_size;

  const float* inputs = (const float*)d_in[0];
  const float* W_in   = (const float*)d_in[1];
  const float* b_in   = (const float*)d_in[2];
  const float* ln1    = (const float*)d_in[3];
  const float* Wqkv   = (const float*)d_in[4];
  const float* Wo     = (const float*)d_in[5];
  const float* ln2    = (const float*)d_in[6];
  const float* W1     = (const float*)d_in[7];
  const float* W2     = (const float*)d_in[8];
  const float* lnf    = (const float*)d_in[9];
  const float* Wout   = (const float*)d_in[10];
  const float* Wqs    = (const float*)d_in[11];
  float* med = (float*)d_out;

  // ---- f32 workspace carve ----
  size_t wofs = 0;
  float* base = (float*)d_ws;
  auto take = [&](size_t n) { float* r = base + wofs; wofs += n; return r; };

  float* normed = take(BSZ * NPAT * PLEN);
  float* mus    = take(BSZ * NPAT);
  float* sigs   = take(BSZ * NPAT);
  float* carry  = take(BSZ * 3);
  float* feat   = take((size_t)BSZ * NPAT * 64);
  float* hdd    = take((size_t)BSZ * NPAT * DMODEL);
  float* xn     = take((size_t)BSZ * NPAT * DMODEL);
  float* hf     = take((size_t)BSZ * NPAT * DMODEL);
  float* qkvb   = take((size_t)BSZ * NPAT * DFF3);
  float* attnb  = take((size_t)BSZ * NPAT * DMODEL);
  float* ffnb   = take((size_t)BSZ * NPAT * DFFN);
  float* cacheK = take((size_t)8 * BSZ * CMAX * DMODEL);
  float* cacheV = take((size_t)8 * BSZ * CMAX * DMODEL);
  float* no_    = take((size_t)BSZ * NPAT * (ODIM * QDIM));
  float* nqs    = take((size_t)BSZ * NPAT * (OSDIM * QDIM));
  float* pfP[2] = { take(BSZ * ODIM * QDIM),  take(BSZ * ODIM * QDIM) };
  float* qsP[2] = { take(BSZ * OSDIM * QDIM), take(BSZ * OSDIM * QDIM) };
  float* arP[2] = { take(BSZ * ODIM * QDIM),  take(BSZ * ODIM * QDIM) };
  float* arx    = take(BSZ * ODIM);
  float* nnorm  = take(BSZ * MPAT * PLEN);
  float* nmus   = take(BSZ * MPAT);
  float* nsigs  = take(BSZ * MPAT);
  unsigned char* isp = (unsigned char*)take(8);
  wofs = (wofs + 7) & ~(size_t)7;                 // 16B-align bf16 arena

  // ---- bf16 transposed-weight arena ----
  const size_t nWin  = (size_t)DMODEL * 64;
  const size_t nWqkv = (size_t)8 * DFF3 * DMODEL;
  const size_t nWo   = (size_t)8 * DMODEL * DMODEL;
  const size_t nW1   = (size_t)8 * DFFN * DMODEL;
  const size_t nW2   = (size_t)8 * DMODEL * DFFN;
  const size_t nWout = (size_t)(ODIM * QDIM) * DMODEL;
  const size_t nWqs  = (size_t)(OSDIM * QDIM) * DMODEL;
  const size_t nBf16 = nWin + nWqkv + nWo + nW1 + nW2 + nWout + nWqs;
  const size_t need  = wofs * sizeof(float) + nBf16 * sizeof(__bf16);
  const bool cached  = (ws_size >= need);

  WTb wt = {};
  if (cached) {
    __bf16* bp = (__bf16*)(base + wofs);
    auto btake = [&](size_t n) { __bf16* r = bp; bp += n; return r; };
    __bf16* tWin  = btake(nWin);
    __bf16* tWqkv = btake(nWqkv);
    __bf16* tWo   = btake(nWo);
    __bf16* tW1   = btake(nW1);
    __bf16* tW2   = btake(nW2);
    __bf16* tWout = btake(nWout);
    __bf16* tWqs  = btake(nWqs);
    wt = { tWin, tWqkv, tWo, tW1, tW2, tWout, tWqs };

    auto convW = [&](const float* W, __bf16* Wt, int K, int N) {
      dim3 g(K / 32, N / 32), b(256);
      convert_wT<<<g, b, 0, stream>>>(W, Wt, K, N);
    };
    convW(W_in, tWin, 64, DMODEL);
    for (int l = 0; l < 8; ++l) {
      convW(Wqkv + (size_t)l * DMODEL * DFF3, tWqkv + (size_t)l * DFF3 * DMODEL, DMODEL, DFF3);
      convW(Wo   + (size_t)l * DMODEL * DMODEL, tWo + (size_t)l * DMODEL * DMODEL, DMODEL, DMODEL);
      convW(W1   + (size_t)l * DMODEL * DFFN, tW1 + (size_t)l * DFFN * DMODEL, DMODEL, DFFN);
      convW(W2   + (size_t)l * DFFN * DMODEL, tW2 + (size_t)l * DMODEL * DFFN, DFFN, DMODEL);
    }
    convW(Wout, tWout, DMODEL, ODIM * QDIM);
    convW(Wqs,  tWqs,  DMODEL, OSDIM * QDIM);
  }

  ispos_k<<<1, 32, 0, stream>>>(inputs, isp);

  for (int pass = 0; pass < 2; ++pass) {
    const float sign = (pass == 0) ? 1.f : -1.f;
    const int MrP = BSZ * NPAT;   // 512
    const int MrA = BSZ * MPAT;   // 128

    // ---- prefill ----
    stats_prefill<<<1, 32, 0, stream>>>(inputs, sign, normed, mus, sigs, carry);
    build_feat<<<(MrP * 64 + 255) / 256, 256, 0, stream>>>(normed, feat, MrP);
    run_model(stream, cached, NPAT, 0, W_in, b_in, ln1, Wqkv, Wo, ln2, W1, W2,
              lnf, wt, feat, hdd, xn, qkvb, attnb, ffnb, cacheK, cacheV, hf);
    gemm(stream, cached, hf, Wout, wt.Wout, nullptr, nullptr, no_,
         MrP, ODIM * QDIM, DMODEL, 0);
    gemm(stream, cached, hf, Wqs, wt.Wqs, nullptr, nullptr, nqs,
         MrP, OSDIM * QDIM, DMODEL, 0);
    revin_rows<<<(BSZ * ODIM * QDIM + 255) / 256, 256, 0, stream>>>(
        no_, mus, sigs, NPAT - 1, NPAT, ODIM * QDIM, pfP[pass]);
    revin_rows<<<(BSZ * OSDIM * QDIM + 255) / 256, 256, 0, stream>>>(
        nqs, mus, sigs, NPAT - 1, NPAT, OSDIM * QDIM, qsP[pass]);

    // ---- single AR step ----
    extract_ar<<<(BSZ * ODIM + 255) / 256, 256, 0, stream>>>(pfP[pass], arx);
    stats_ar<<<1, 32, 0, stream>>>(arx, carry, nnorm, nmus, nsigs);
    build_feat<<<(MrA * 64 + 255) / 256, 256, 0, stream>>>(nnorm, feat, MrA);
    run_model(stream, cached, MPAT, NPAT, W_in, b_in, ln1, Wqkv, Wo, ln2, W1, W2,
              lnf, wt, feat, hdd, xn, qkvb, attnb, ffnb, cacheK, cacheV, hf);
    gemm(stream, cached, hf, Wout, wt.Wout, nullptr, nullptr, no_,
         MrA, ODIM * QDIM, DMODEL, 0);
    revin_rows<<<(BSZ * ODIM * QDIM + 255) / 256, 256, 0, stream>>>(
        no_, nmus, nsigs, MPAT - 1, MPAT, ODIM * QDIM, arP[pass]);
  }

  combine_k<<<(BSZ * HOR + 255) / 256, 256, 0, stream>>>(
      pfP[0], arP[0], qsP[0], pfP[1], arP[1], qsP[1], isp, med);
}
